// Inner_74277164417609
// MI455X (gfx1250) — matrix operand order repaired and checked by reading, not compiled, not run on hardware
//
#include <hip/hip_runtime.h>
#include <hip/hip_bf16.h>

#define NATOMS 30000
#define NEDGES 480000
#define NGRAPH 64
#define HID    128
#define NFILT  128
#define NGAUSS 50
#define NLAYER 6
#define DEH    256
#define LOG2F_ 0.6931471805599453f

typedef _Float16 h16_t;
typedef __attribute__((ext_vector_type(2)))  _Float16 h2v;
typedef __attribute__((ext_vector_type(16))) _Float16 v16h;
typedef __attribute__((ext_vector_type(8)))  float    v8f;

// fast 1-instruction reciprocal (v_rcp_f32) -- avoids IEEE div expansion
__device__ __forceinline__ float frcp(float x) { return __builtin_amdgcn_rcpf(x); }

__device__ __forceinline__ float sspf(float x) {
  return (x > 15.f) ? (x - LOG2F_) : (__logf(1.f + __expf(x)) - LOG2F_);
}
__device__ __forceinline__ float sigm(float x) { return frcp(1.f + __expf(-x)); }

// A-fragment (16x32 f16, M x K) per CDNA5 ISA 7.12.2:
// lane 0-15 -> M=lane, K pairs {0,1},{2,3},{4,5},{6,7},{16,17},{18,19},{20,21},{22,23}
// lane 16-31 -> same M, K base +8 (and +8 in upper group)
__device__ __forceinline__ v16h ldA(const h16_t* tile, int ld, int lane, int k0) {
  const h16_t* rp = tile + (size_t)(lane & 15) * ld + k0 + ((lane & 16) ? 8 : 0);
  v16h a;
#pragma unroll
  for (int v = 0; v < 8; ++v) {
    const int kk = (v < 4) ? (2 * v) : (16 + 2 * (v - 4));
    h2v p = *(const h2v*)(rp + kk);
    a[2 * v] = p.x; a[2 * v + 1] = p.y;
  }
  return a;
}
// B-fragment (32x16 f16, K x N): N = lane&15, K in same order as A fragment
__device__ __forceinline__ v16h ldB(const h16_t* B, int ld, int lane, int k0, int n0) {
  const h16_t* cp = B + (size_t)(k0 + ((lane & 16) ? 8 : 0)) * ld + n0 + (lane & 15);
  v16h b;
#pragma unroll
  for (int s = 0; s < 16; ++s) {
    const int kk = (s < 8) ? s : (16 + (s - 8));
    b[s] = cp[(size_t)kk * ld];
  }
  return b;
}
#define WMMA_F16(a,b,c) __builtin_amdgcn_wmma_f32_16x16x32_f16(false,(a),false,(b),(short)0,(c),false,false)

// ---------------- weight conversion to f16 [K][N] (padded) ----------------
__global__ void k_convert(const float* de_w2, const float* mlp1_w, const float* mlp2_w,
                          const float* lin1c_w, const float* lin2c_w, const float* linb_w,
                          h16_t* Bde, h16_t* B1, h16_t* B2, h16_t* Bl1, h16_t* Bl2, h16_t* Bb) {
  int tid = blockIdx.x * blockDim.x + threadIdx.x;
  if (tid < DEH * 64) {                       // Bde[k=256][n=64pad]
    int k = tid >> 6, n = tid & 63;
    Bde[tid] = (n < NGAUSS) ? (h16_t)de_w2[n * DEH + k] : (h16_t)0.f;
  }
  if (tid < NLAYER * 64 * NFILT) {            // B1[i][k=64pad][n=128]
    int i = tid / (64 * NFILT), r = tid % (64 * NFILT);
    int k = r >> 7, n = r & 127;
    B1[tid] = (k < NGAUSS) ? (h16_t)mlp1_w[(i * NFILT + n) * NGAUSS + k] : (h16_t)0.f;
  }
  if (tid < NLAYER * NFILT * NFILT) {         // 128x128 families: B[k][n] = W[n][k]
    int i = tid / (NFILT * NFILT), r = tid % (NFILT * NFILT);
    int k = r >> 7, n = r & 127;
    int src = (i * 128 + n) * 128 + k;
    B2[tid]  = (h16_t)mlp2_w[src];
    Bl1[tid] = (h16_t)lin1c_w[src];
    Bl2[tid] = (h16_t)lin2c_w[src];
    Bb[tid]  = (h16_t)linb_w[src];
  }
}

// ---------------- embedding gather (padding_idx 0) ----------------
__global__ void k_embed(const int* z, const float* emb, float* h, h16_t* h16o) {
  int tid = blockIdx.x * blockDim.x + threadIdx.x;
  if (tid >= NATOMS * HID) return;
  int n = tid & 127, a = tid >> 7;
  int zz = z[a];
  float v = (zz == 0) ? 0.f : emb[zz * HID + n];
  h[tid] = v; h16o[tid] = (h16_t)v;
}

// ---------------- distance expansion: Linear(1,256)->LN->SiLU->WMMA(256->50)->sigmoid ----
__global__ void k_distexp(const float* pos, const int* ei,
                          const float* de_w1, const float* de_b1,
                          const float* ln_g, const float* ln_b,
                          const h16_t* Bde, const float* de_b2,
                          h16_t* edge_attr) {
  __shared__ h16_t stage[4][16][DEH];
  const int lane = threadIdx.x & 31, wid = threadIdx.x >> 5;
  const int e0 = (blockIdx.x * 4 + wid) * 16;
  const int* row = ei; const int* col = ei + NEDGES;
  {
    int e = e0 + (lane >> 1);
    int j0 = (lane & 1) * 128;           // lane pair covers 256 cols of one edge
    int rs = row[e], cs = col[e];
    float dx = pos[rs * 3 + 0] - pos[cs * 3 + 0];
    float dy = pos[rs * 3 + 1] - pos[cs * 3 + 1];
    float dz = pos[rs * 3 + 2] - pos[cs * 3 + 2];
    float d  = sqrtf(dx * dx + dy * dy + dz * dz);
    float dn = (d - 2.7554f) * (1.0f / 1.1664f);
    float s1 = 0.f, s2 = 0.f;
#pragma unroll 4
    for (int j = 0; j < 128; ++j) {
      float t = fmaf(dn, de_w1[j0 + j], de_b1[j0 + j]);
      s1 += t; s2 += t * t;
    }
    s1 += __shfl_xor(s1, 1); s2 += __shfl_xor(s2, 1);
    float mu = s1 * (1.f / 256.f);
    float var = s2 * (1.f / 256.f) - mu * mu;
    float rstd = rsqrtf(var + 1e-5f);
#pragma unroll 4
    for (int j = 0; j < 128; ++j) {
      int jj = j0 + j;
      float t = fmaf(dn, de_w1[jj], de_b1[jj]);
      t = (t - mu) * rstd * ln_g[jj] + ln_b[jj];
      t = t * sigm(t);                   // SiLU (fast rcp path)
      stage[wid][lane >> 1][jj] = (h16_t)t;
    }
  }
  __syncthreads();
  v16h aA[8];
#pragma unroll
  for (int ks = 0; ks < 8; ++ks) aA[ks] = ldA(&stage[wid][0][0], DEH, lane, ks * 32);
#pragma unroll
  for (int nt = 0; nt < 4; ++nt) {
    v8f acc = {};
#pragma unroll
    for (int ks = 0; ks < 8; ++ks)
      acc = WMMA_F16(aA[ks], ldB(Bde, 64, lane, ks * 32, nt * 16), acc);
    int n = nt * 16 + (lane & 15);
    float bias = (n < NGAUSS) ? de_b2[n] : 0.f;
#pragma unroll
    for (int r = 0; r < 8; ++r) {
      int m = r + ((lane & 16) ? 8 : 0);
      edge_attr[(size_t)(e0 + m) * 64 + n] = (h16_t)sigm(acc[r] + bias);
    }
  }
}

// ---------------- x1 = h @ lin1c^T (pure WMMA) ----------------
__global__ void k_x1(const h16_t* hA, const h16_t* Bl1, float* x1) {
  const int lane = threadIdx.x & 31, wid = threadIdx.x >> 5;
  const int tile = blockIdx.x * 4 + wid;
  if (tile >= NATOMS / 16) return;
  const int m0 = tile * 16;
  v16h aA[4];
#pragma unroll
  for (int ks = 0; ks < 4; ++ks) aA[ks] = ldA(hA + (size_t)m0 * HID, HID, lane, ks * 32);
#pragma unroll
  for (int nt = 0; nt < 8; ++nt) {
    v8f acc = {};
#pragma unroll
    for (int ks = 0; ks < 4; ++ks)
      acc = WMMA_F16(aA[ks], ldB(Bl1, NFILT, lane, ks * 32, nt * 16), acc);
    int n = nt * 16 + (lane & 15);
#pragma unroll
    for (int r = 0; r < 8; ++r) {
      int m = r + ((lane & 16) ? 8 : 0);
      x1[(size_t)(m0 + m) * NFILT + n] = acc[r];
    }
  }
}

// ------- fused edge filter: GEMM1 -> ssp -> GEMM2 -> gather-multiply -> scatter-atomic ----
__global__ void k_edge(const h16_t* edge_attr, const int* ei,
                       const h16_t* B1, const float* b1,
                       const h16_t* B2, const float* b2,
                       const float* x1, float* agg) {
  __shared__ h16_t stage[4][16][NFILT];
  const int lane = threadIdx.x & 31, wid = threadIdx.x >> 5;
  const int e0 = (blockIdx.x * 4 + wid) * 16;
  // GEMM1: [16,64] x [64,128]
  v16h aA0 = ldA(edge_attr + (size_t)e0 * 64, 64, lane, 0);
  v16h aA1 = ldA(edge_attr + (size_t)e0 * 64, 64, lane, 32);
#pragma unroll
  for (int nt = 0; nt < 8; ++nt) {
    v8f acc = {};
    acc = WMMA_F16(aA0, ldB(B1, NFILT, lane,  0, nt * 16), acc);
    acc = WMMA_F16(aA1, ldB(B1, NFILT, lane, 32, nt * 16), acc);
    int n = nt * 16 + (lane & 15);
    float bias = b1[n];
#pragma unroll
    for (int r = 0; r < 8; ++r) {
      int m = r + ((lane & 16) ? 8 : 0);
      stage[wid][m][n] = (h16_t)sspf(acc[r] + bias);
    }
  }
  __syncthreads();
  // GEMM2: [16,128] x [128,128] from LDS staging
  v16h aB[4];
#pragma unroll
  for (int ks = 0; ks < 4; ++ks) aB[ks] = ldA(&stage[wid][0][0], NFILT, lane, ks * 32);
  const int* row = ei; const int* col = ei + NEDGES;
  int rs[8], cs[8];
#pragma unroll
  for (int r = 0; r < 8; ++r) {
    int m = r + ((lane & 16) ? 8 : 0);
    rs[r] = row[e0 + m]; cs[r] = col[e0 + m];
  }
#pragma unroll
  for (int nt = 0; nt < 8; ++nt) {
    v8f acc = {};
#pragma unroll
    for (int ks = 0; ks < 4; ++ks)
      acc = WMMA_F16(aB[ks], ldB(B2, NFILT, lane, ks * 32, nt * 16), acc);
    int n = nt * 16 + (lane & 15);
    float bias = b2[n];
#pragma unroll
    for (int r = 0; r < 8; ++r) {
      float W  = acc[r] + bias;                             // filter value (never hits HBM)
      float xv = x1[(size_t)rs[r] * NFILT + n];             // L2-resident gather
      atomicAdd(&agg[(size_t)cs[r] * NFILT + n], W * xv);   // L2-resident scatter-add
    }
  }
}

// ------- fused node update: agg@lin2^T+b -> ssp -> @linb^T+b -> residual into h/h16 ----
__global__ void k_node(const float* agg, const h16_t* Bl2, const float* bl2,
                       const h16_t* Bb, const float* bb,
                       float* h, h16_t* h16o) {
  __shared__ h16_t stage[4][16][HID];
  const int lane = threadIdx.x & 31, wid = threadIdx.x >> 5;
  const int tile = blockIdx.x * 4 + wid;
  const bool active = tile < (NATOMS / 16);
  const int m0 = tile * 16;
  if (active) {
#pragma unroll 8
    for (int t = 0; t < 64; ++t) {
      int flat = lane * 64 + t;
      int m = flat >> 7, n = flat & 127;
      stage[wid][m][n] = (h16_t)agg[(size_t)(m0 + m) * HID + n];
    }
  }
  __syncthreads();
  v8f acc[8];
  if (active) {
    v16h aA[4];
#pragma unroll
    for (int ks = 0; ks < 4; ++ks) aA[ks] = ldA(&stage[wid][0][0], HID, lane, ks * 32);
#pragma unroll
    for (int nt = 0; nt < 8; ++nt) {
      v8f a = {};
#pragma unroll
      for (int ks = 0; ks < 4; ++ks)
        a = WMMA_F16(aA[ks], ldB(Bl2, HID, lane, ks * 32, nt * 16), a);
      acc[nt] = a;
    }
  }
  __syncthreads();
  if (active) {
#pragma unroll
    for (int nt = 0; nt < 8; ++nt) {
      int n = nt * 16 + (lane & 15);
      float bias = bl2[n];
#pragma unroll
      for (int r = 0; r < 8; ++r) {
        int m = r + ((lane & 16) ? 8 : 0);
        stage[wid][m][n] = (h16_t)sspf(acc[nt][r] + bias);
      }
    }
  }
  __syncthreads();
  if (active) {
    v16h aA[4];
#pragma unroll
    for (int ks = 0; ks < 4; ++ks) aA[ks] = ldA(&stage[wid][0][0], HID, lane, ks * 32);
#pragma unroll
    for (int nt = 0; nt < 8; ++nt) {
      v8f a = {};
#pragma unroll
      for (int ks = 0; ks < 4; ++ks)
        a = WMMA_F16(aA[ks], ldB(Bb, HID, lane, ks * 32, nt * 16), a);
      int n = nt * 16 + (lane & 15);
      float bias = bb[n];
#pragma unroll
      for (int r = 0; r < 8; ++r) {
        int m = r + ((lane & 16) ? 8 : 0);
        size_t idx = (size_t)(m0 + m) * HID + n;
        float nh = h[idx] + a[r] + bias;
        h[idx] = nh; h16o[idx] = (h16_t)nh;
      }
    }
  }
}

// ---------------- readout head + per-graph segment sum ----------------
__global__ void k_readout(const float* h, const int* batch,
                          const float* w1, const float* b1v,
                          const float* w2, const float* b2v,
                          float* out) {
  __shared__ float red[64];
  int a = blockIdx.x, j = threadIdx.x;
  const float* hr = h + (size_t)a * HID;
  float dot = 0.f;
#pragma unroll 4
  for (int k = 0; k < HID; ++k) dot = fmaf(hr[k], w1[j * HID + k], dot);
  red[j] = sspf(dot + b1v[j]) * w2[j];
  __syncthreads();
  for (int s = 32; s > 0; s >>= 1) { if (j < s) red[j] += red[j + s]; __syncthreads(); }
  if (j == 0) atomicAdd(&out[batch[a]], red[0] + b2v[0]);
}

extern "C" void kernel_launch(void* const* d_in, const int* in_sizes, int n_in,
                              void* d_out, int out_size, void* d_ws, size_t ws_size,
                              hipStream_t stream) {
  (void)in_sizes; (void)n_in; (void)out_size; (void)ws_size;
  const int*   z      = (const int*)d_in[0];
  const float* pos    = (const float*)d_in[1];
  const int*   batch  = (const int*)d_in[2];
  const int*   ei     = (const int*)d_in[3];
  const float* emb    = (const float*)d_in[4];
  const float* de_w1  = (const float*)d_in[5];
  const float* de_b1  = (const float*)d_in[6];
  const float* ln_g   = (const float*)d_in[7];
  const float* ln_b   = (const float*)d_in[8];
  const float* de_w2  = (const float*)d_in[9];
  const float* de_b2  = (const float*)d_in[10];
  const float* mlp1_w = (const float*)d_in[11];
  const float* mlp1_b = (const float*)d_in[12];
  const float* mlp2_w = (const float*)d_in[13];
  const float* mlp2_b = (const float*)d_in[14];
  const float* lin1c_w= (const float*)d_in[15];
  const float* lin2c_w= (const float*)d_in[16];
  const float* lin2c_b= (const float*)d_in[17];
  const float* linb_w = (const float*)d_in[18];
  const float* linb_b = (const float*)d_in[19];
  const float* out1_w = (const float*)d_in[20];
  const float* out1_b = (const float*)d_in[21];
  const float* out2_w = (const float*)d_in[22];
  const float* out2_b = (const float*)d_in[23];
  float* out = (float*)d_out;

  char* ws = (char*)d_ws;
  size_t off = 0;
  auto alloc = [&](size_t bytes) { void* p = ws + off; off += (bytes + 255) & ~(size_t)255; return p; };
  float* h    = (float*)alloc((size_t)NATOMS * HID * 4);
  float* x1   = (float*)alloc((size_t)NATOMS * HID * 4);
  float* agg  = (float*)alloc((size_t)NATOMS * HID * 4);
  h16_t* h16b = (h16_t*)alloc((size_t)NATOMS * HID * 2);
  h16_t* ea   = (h16_t*)alloc((size_t)NEDGES * 64 * 2);
  h16_t* Bde  = (h16_t*)alloc((size_t)DEH * 64 * 2);
  h16_t* B1   = (h16_t*)alloc((size_t)NLAYER * 64 * NFILT * 2);
  h16_t* B2   = (h16_t*)alloc((size_t)NLAYER * NFILT * NFILT * 2);
  h16_t* Bl1  = (h16_t*)alloc((size_t)NLAYER * NFILT * NFILT * 2);
  h16_t* Bl2  = (h16_t*)alloc((size_t)NLAYER * NFILT * NFILT * 2);
  h16_t* Bb   = (h16_t*)alloc((size_t)NLAYER * NFILT * NFILT * 2);

  k_convert<<<384, 256, 0, stream>>>(de_w2, mlp1_w, mlp2_w, lin1c_w, lin2c_w, linb_w,
                                     Bde, B1, B2, Bl1, Bl2, Bb);
  k_embed<<<(NATOMS * HID + 255) / 256, 256, 0, stream>>>(z, emb, h, h16b);
  k_distexp<<<NEDGES / 64, 128, 0, stream>>>(pos, ei, de_w1, de_b1, ln_g, ln_b, Bde, de_b2, ea);

  const int ntile = NATOMS / 16;        // 1875 exact
  const int nblk  = (ntile + 3) / 4;    // 469
  for (int i = 0; i < NLAYER; ++i) {
    hipMemsetAsync(agg, 0, (size_t)NATOMS * HID * 4, stream);
    k_x1<<<nblk, 128, 0, stream>>>(h16b, Bl1 + (size_t)i * NFILT * NFILT, x1);
    k_edge<<<NEDGES / 64, 128, 0, stream>>>(ea, ei,
        B1 + (size_t)i * 64 * NFILT, mlp1_b + (size_t)i * NFILT,
        B2 + (size_t)i * NFILT * NFILT, mlp2_b + (size_t)i * NFILT,
        x1, agg);
    k_node<<<nblk, 128, 0, stream>>>(agg,
        Bl2 + (size_t)i * NFILT * NFILT, lin2c_b + (size_t)i * HID,
        Bb  + (size_t)i * NFILT * NFILT, linb_b  + (size_t)i * HID,
        h, h16b);
  }
  hipMemsetAsync(out, 0, NGRAPH * sizeof(float), stream);
  k_readout<<<NATOMS, 64, 0, stream>>>(h, batch, out1_w, out1_b, out2_w, out2_b, out);
}
